// ViT_45105746542811
// MI455X (gfx1250) — compile-verified, matrix operand order checked by reading, not run on hardware
//
#include <hip/hip_runtime.h>
#include <hip/hip_bf16.h>

// ---------------------------------------------------------------------------
// ViT-Base forward for MI455X (gfx1250, wave32, WMMA).
// GEMMs: v_wmma_f32_16x16x32_f16, f32 accumulate, f16 operands, weights
// pre-transposed to [N][K].  Block tile 128x256x32, 8 waves, wave tile 64x64
// (4x4 WMMA tiles).  Double-buffered LDS fed by GLOBAL_LOAD_ASYNC_TO_LDS
// (ASYNCcnt), steady-state loop branch-free (last iteration peeled).
// Residual stream fp32; LN/softmax/GELU fp32.
// ---------------------------------------------------------------------------

typedef _Float16 h16 __attribute__((ext_vector_type(16)));
typedef _Float16 h8  __attribute__((ext_vector_type(8)));
typedef float    f8  __attribute__((ext_vector_type(8)));
typedef int      v4i __attribute__((ext_vector_type(4)));
typedef __attribute__((address_space(1))) v4i* as1_v4i;
typedef __attribute__((address_space(3))) v4i* as3_v4i;

#if defined(__has_builtin)
#if __has_builtin(__builtin_amdgcn_global_load_async_to_lds_b128)
#define HAVE_ASYNC_LDS 1
#endif
#if __has_builtin(__builtin_amdgcn_s_wait_asynccnt)
#define HAVE_WAIT_ASYNC 1
#endif
#endif

// Model constants
constexpr int D_    = 768;
constexpr int HEADS = 8;
constexpr int DHD   = 64;
constexpr int MLP_  = 2048;
constexpr int NTOK  = 1025;
constexpr int BB    = 4;
constexpr int MT    = BB * NTOK;     // 4100
constexpr int MTP   = 4224;          // 33*128
constexpr int SP    = 1152;          // padded seq (9*128)
constexpr int NPAT  = 4096;
constexpr int KPAT  = 256;
constexpr int QKVN  = 3 * HEADS * DHD;  // 1536
constexpr int HD    = HEADS * DHD;      // 512
constexpr int DEPTH = 12;

__device__ __forceinline__ f8 f8zero() {
  f8 r;
#pragma unroll
  for (int i = 0; i < 8; ++i) r[i] = 0.0f;
  return r;
}

template <int N>
__device__ __forceinline__ void wait_async() {
#if defined(HAVE_WAIT_ASYNC)
  __builtin_amdgcn_s_wait_asynccnt(N);
#else
  asm volatile("s_wait_asynccnt %0" ::"n"(N) : "memory");
#endif
}

__device__ __forceinline__ void cp16_async(_Float16* l, const _Float16* g) {
#if defined(HAVE_ASYNC_LDS)
  __builtin_amdgcn_global_load_async_to_lds_b128((as1_v4i)g, (as3_v4i)l, 0, 0);
#else
  (void)l; (void)g;
#endif
}

// Load one 16x32 f16 WMMA fragment from an LDS tile (row stride 40 halfs).
// Lane layout per CDNA5 ISA 7.12.2: m/n = lane&15, half = lane>>4,
// elems 0..7 = K[half*8..+7], elems 8..15 = K[16+half*8..+7].
__device__ __forceinline__ h16 ldfrag(const _Float16* s, int row, int half) {
  const h8 lo = *(const h8*)(s + row * 40 + half * 8);
  const h8 hi = *(const h8*)(s + row * 40 + 16 + half * 8);
  return __builtin_shufflevector(lo, hi, 0, 1, 2, 3, 4, 5, 6, 7,
                                 8, 9, 10, 11, 12, 13, 14, 15);
}

struct GemmP {
  const _Float16* A;   // [Mpad][lda], rows fully allocated (no guard needed)
  const _Float16* B;   // [nbRows][ldb] = W^T / K / V^T
  float*          of;
  _Float16*       oh;
  _Float16*       oh2;
  _Float16*       oh3;
  const float*    bias;
  const float*    aux;
  int lda, ldb, K, nbRows, mValid, ldo;
  long long aBatch, bBatch, oBatch;
};

// EPI: 0=patch-embed  1=qkv-scatter  2=scores  3=attnV  4=bias+GELU  5=residual
template <int EPI>
__global__ __launch_bounds__(256) void gemm_wmma(GemmP p) {
  __shared__ __align__(16) _Float16 sA[2][128 * 40];
  __shared__ __align__(16) _Float16 sB[2][256 * 40];

  const int tid  = threadIdx.x;
  const int lane = tid & 31;
  const int wave = tid >> 5;   // 8 waves
  const int wm   = wave >> 2;  // 0..1 -> 64 rows
  const int wn   = wave & 3;   // 0..3 -> 64 cols
  const int hl   = lane >> 4;
  const int ml   = lane & 15;
  const int z    = blockIdx.z;
  const int m0   = blockIdx.y * 128;
  const int n0   = blockIdx.x * 256;

  const _Float16* A  = p.A + (size_t)z * p.aBatch;
  const _Float16* Bm = p.B + (size_t)z * p.bBatch;

  // Per-thread copy slots: A = 2 chunks (128 rows), B = 4 chunks (256 rows).
  // B rows are clamped (branchless) -- out-of-range rows only feed output
  // columns that the epilogue never stores.
  const int lr = tid >> 2;        // 0..63
  const int lc = (tid & 3) * 8;   // 0,8,16,24
  const _Float16* gA[2];
  const _Float16* gB[4];
#pragma unroll
  for (int it = 0; it < 2; ++it)
    gA[it] = A + (size_t)(m0 + lr + it * 64) * p.lda + lc;
#pragma unroll
  for (int it = 0; it < 4; ++it) {
    int rb = n0 + lr + it * 64;
    rb = rb < p.nbRows ? rb : p.nbRows - 1;
    gB[it] = Bm + (size_t)rb * p.ldb + lc;
  }

  f8 acc[4][4];
#pragma unroll
  for (int mt = 0; mt < 4; ++mt)
#pragma unroll
    for (int nt = 0; nt < 4; ++nt) acc[mt][nt] = f8zero();

  auto mma_tile = [&](int sel) {
    h16 af[4], bf[4];
#pragma unroll
    for (int mt = 0; mt < 4; ++mt)
      af[mt] = ldfrag(&sA[sel][0], wm * 64 + mt * 16 + ml, hl);
#pragma unroll
    for (int nt = 0; nt < 4; ++nt)
      bf[nt] = ldfrag(&sB[sel][0], wn * 64 + nt * 16 + ml, hl);
#pragma unroll
    for (int mt = 0; mt < 4; ++mt)
#pragma unroll
      for (int nt = 0; nt < 4; ++nt)
        acc[mt][nt] = __builtin_amdgcn_wmma_f32_16x16x32_f16(
            false, af[mt], false, bf[nt], (short)0, acc[mt][nt], false, false);
  };

  int sel = 0;
#if defined(HAVE_ASYNC_LDS)
  // Async-to-LDS double-buffered pipeline: each wave issues exactly 6 async
  // b128 ops per tile; async loads complete in order, so after issuing tile
  // k+1, s_wait_asynccnt<=6 guarantees tile k is resident.  Last iteration is
  // peeled so the steady-state loop is branch-free.
  auto copy_tile = [&](int s2, int kk) {
    _Float16* dA = &sA[s2][0];
    _Float16* dB = &sB[s2][0];
#pragma unroll
    for (int it = 0; it < 2; ++it)
      cp16_async(dA + (lr + it * 64) * 40 + lc, gA[it] + kk);
#pragma unroll
    for (int it = 0; it < 4; ++it)
      cp16_async(dB + (lr + it * 64) * 40 + lc, gB[it] + kk);
  };
  copy_tile(0, 0);
  for (int kk = 0; kk + 32 < p.K; kk += 32) {
    copy_tile(sel ^ 1, kk + 32);
    wait_async<6>();
    __syncthreads();
    mma_tile(sel);
    __syncthreads();
    sel ^= 1;
  }
  wait_async<0>();
  __syncthreads();
  mma_tile(sel);
#else
  // Fallback: register-staged software pipeline (load k+1 while computing k).
  h8 stA[2], stB[4];
  auto ld_tile = [&](int kk) {
#pragma unroll
    for (int it = 0; it < 2; ++it) stA[it] = *(const h8*)(gA[it] + kk);
#pragma unroll
    for (int it = 0; it < 4; ++it) stB[it] = *(const h8*)(gB[it] + kk);
  };
  auto st_tile = [&](int s2) {
#pragma unroll
    for (int it = 0; it < 2; ++it)
      *(h8*)(&sA[s2][(lr + it * 64) * 40 + lc]) = stA[it];
#pragma unroll
    for (int it = 0; it < 4; ++it)
      *(h8*)(&sB[s2][(lr + it * 64) * 40 + lc]) = stB[it];
  };
  ld_tile(0);
  for (int kk = 0; kk + 32 < p.K; kk += 32) {
    st_tile(sel);
    __syncthreads();
    ld_tile(kk + 32);
    mma_tile(sel);
    __syncthreads();
    sel ^= 1;
  }
  st_tile(sel);
  __syncthreads();
  mma_tile(sel);
#endif

  // Epilogue.  C layout: n = lane&15, m = e + 8*(lane>>4) within each 16x16.
#pragma unroll
  for (int mt = 0; mt < 4; ++mt) {
#pragma unroll
    for (int nt = 0; nt < 4; ++nt) {
      const int n  = n0 + wn * 64 + nt * 16 + ml;
      const int mb = m0 + wm * 64 + mt * 16 + hl * 8;
#pragma unroll
      for (int e = 0; e < 8; ++e) {
        const int   m = mb + e;
        const float v = acc[mt][nt][e];
        if constexpr (EPI == 0) {  // patch embed -> t (fp32) + conv_b + pos
          const int b = m >> 10, hw = m & 1023;
          const size_t row = (size_t)(b * NTOK + 1 + hw);
          p.of[row * D_ + n] = v + p.bias[n] + p.aux[(size_t)(1 + hw) * D_ + n];
        } else if constexpr (EPI == 1) {  // qkv scatter (q,k row / v transposed)
          if (m < p.mValid) {
            const int b = m / NTOK, nn = m % NTOK;
            const int h = n / 192, r = (n / 64) % 3, dh = n & 63;
            const int bh = b * HEADS + h;
            const _Float16 hv = (_Float16)v;
            if (r == 0)      p.oh [((size_t)bh * SP + nn) * DHD + dh] = hv;
            else if (r == 1) p.oh2[((size_t)bh * SP + nn) * DHD + dh] = hv;
            else             p.oh3[((size_t)bh * DHD + dh) * SP + nn] = hv;
          }
        } else if constexpr (EPI == 2) {  // attention scores -> f16
          if (n < SP)
            p.oh[(size_t)z * p.oBatch + (size_t)m * SP + n] = (_Float16)v;
        } else if constexpr (EPI == 3) {  // attn @ V -> merged heads (f16)
          if (m < NTOK && n < DHD) {
            const int b = z >> 3, h = z & 7;
            p.oh[(size_t)(b * NTOK + m) * HD + h * DHD + n] = (_Float16)v;
          }
        } else if constexpr (EPI == 4) {  // bias + exact GELU -> f16
          if (m < p.mValid) {
            const float t = v + p.bias[n];
            const float g = 0.5f * t * (1.0f + erff(t * 0.70710678118654752f));
            p.oh[(size_t)m * p.ldo + n] = (_Float16)g;
          }
        } else if constexpr (EPI == 5) {  // residual accumulate into t (fp32)
          if (m < p.mValid) p.of[(size_t)m * p.ldo + n] += v + p.bias[n];
        }
      }
    }
  }
}

// ------------------------- elementwise / row kernels -----------------------

template <bool OUT16>
__global__ __launch_bounds__(256) void ln_kernel(const float* __restrict__ x,
                                                 const float* __restrict__ s,
                                                 const float* __restrict__ b,
                                                 float* of, _Float16* oh) {
  __shared__ float red[256];
  const int row = blockIdx.x, tid = threadIdx.x;
  const float* xr = x + (size_t)row * D_;
  float xv[3];
#pragma unroll
  for (int i = 0; i < 3; ++i) xv[i] = xr[tid + i * 256];
  red[tid] = xv[0] + xv[1] + xv[2];
  __syncthreads();
  for (int st = 128; st; st >>= 1) {
    if (tid < st) red[tid] += red[tid + st];
    __syncthreads();
  }
  const float mean = red[0] * (1.0f / (float)D_);
  __syncthreads();
  float sq = 0.0f;
#pragma unroll
  for (int i = 0; i < 3; ++i) {
    const float d = xv[i] - mean;
    sq += d * d;
  }
  red[tid] = sq;
  __syncthreads();
  for (int st = 128; st; st >>= 1) {
    if (tid < st) red[tid] += red[tid + st];
    __syncthreads();
  }
  const float rs = rsqrtf(red[0] * (1.0f / (float)D_) + 1e-5f);
#pragma unroll
  for (int i = 0; i < 3; ++i) {
    const int c = tid + i * 256;
    const float o = (xv[i] - mean) * rs * s[c] + b[c];
    if constexpr (OUT16) oh[(size_t)row * D_ + c] = (_Float16)o;
    else                 of[(size_t)row * D_ + c] = o;
  }
}

// in-place softmax over f16 score rows; cols >= 1025 forced to zero
__global__ __launch_bounds__(256) void softmax_kernel(_Float16* probs) {
  __shared__ float red[256];
  const int row = blockIdx.x;  // 0..1024
  const int z   = blockIdx.y;  // 0..31
  const int tid = threadIdx.x;
  _Float16* base = probs + ((size_t)z * SP + row) * SP;
  float lv[5];
  int   cnt = 0;
  float mx = -3.4e38f;
  for (int j = tid; j < SP; j += 256) {
    const float s = (j < NTOK) ? (float)base[j] * 0.125f : -3.4e38f;
    lv[cnt++] = s;
    mx = fmaxf(mx, s);
  }
  red[tid] = mx;
  __syncthreads();
  for (int st = 128; st; st >>= 1) {
    if (tid < st) red[tid] = fmaxf(red[tid], red[tid + st]);
    __syncthreads();
  }
  const float rmax = red[0];
  __syncthreads();
  float sum = 0.0f;
  cnt = 0;
  for (int j = tid; j < SP; j += 256) {
    const float e = (j < NTOK) ? expf(lv[cnt] - rmax) : 0.0f;
    lv[cnt++] = e;
    sum += e;
  }
  red[tid] = sum;
  __syncthreads();
  for (int st = 128; st; st >>= 1) {
    if (tid < st) red[tid] += red[tid + st];
    __syncthreads();
  }
  const float inv = 1.0f / red[0];
  cnt = 0;
  for (int j = tid; j < SP; j += 256) base[j] = (_Float16)(lv[cnt++] * inv);
}

// fp32 [K][N] -> f16 [N][K]  (weight transpose + downconvert)
__global__ void tr_kernel(const float* __restrict__ src,
                          _Float16* __restrict__ dst, int K, int N) {
  const long long i = (long long)blockIdx.x * 256 + threadIdx.x;
  if (i < (long long)K * N) {
    const int k = (int)(i / N), n = (int)(i % N);
    dst[(size_t)n * K + k] = (_Float16)src[i];
  }
}

__global__ void cvt_kernel(const float* __restrict__ src,
                           _Float16* __restrict__ dst, long long n) {
  const long long i = (long long)blockIdx.x * 256 + threadIdx.x;
  if (i < n) dst[i] = (_Float16)src[i];
}

// x [4][1][512][512] -> peA [4096][256] f16
__global__ void patchify_kernel(const float* __restrict__ x,
                                _Float16* __restrict__ pA) {
  const long long i = (long long)blockIdx.x * 256 + threadIdx.x;
  if (i < (long long)NPAT * KPAT) {
    const int m = (int)(i >> 8), kk = (int)(i & 255);
    const int b = m >> 10, hw = m & 1023;
    const int gy = hw >> 5, gx = hw & 31;
    const int pp = kk >> 4, qq = kk & 15;
    pA[i] = (_Float16)
        x[(size_t)b * 512 * 512 + (size_t)(gy * 16 + pp) * 512 + gx * 16 + qq];
  }
}

__global__ void init_tok_kernel(const float* __restrict__ cls,
                                const float* __restrict__ pos,
                                float* __restrict__ t) {
  const int i = blockIdx.x * 256 + threadIdx.x;
  if (i < BB * D_) {
    const int b = i / D_, d = i % D_;
    t[(size_t)(b * NTOK) * D_ + d] = cls[d] + pos[d];
  }
}

__global__ void zero_kernel(uint4* __restrict__ p, long long n) {
  const long long i = (long long)blockIdx.x * 256 + threadIdx.x;
  if (i < n) p[i] = make_uint4(0u, 0u, 0u, 0u);
}

// ---------------------------------------------------------------------------

static inline int cdiv(long long a, long long b) { return (int)((a + b - 1) / b); }

extern "C" void kernel_launch(void* const* d_in, const int* in_sizes, int n_in,
                              void* d_out, int out_size, void* d_ws,
                              size_t ws_size, hipStream_t stream) {
  (void)in_sizes; (void)n_in; (void)out_size; (void)ws_size;
  const float* x       = (const float*)d_in[0];
  const float* conv_w  = (const float*)d_in[1];
  const float* conv_b  = (const float*)d_in[2];
  const float* cls_tok = (const float*)d_in[3];
  const float* pos_emb = (const float*)d_in[4];
  const float* qkv_w   = (const float*)d_in[5];
  const float* merge_w = (const float*)d_in[6];
  const float* merge_b = (const float*)d_in[7];
  const float* ln1_s   = (const float*)d_in[8];
  const float* ln1_b   = (const float*)d_in[9];
  const float* ln2_s   = (const float*)d_in[10];
  const float* ln2_b   = (const float*)d_in[11];
  const float* ffn_w1  = (const float*)d_in[12];
  const float* ffn_b1  = (const float*)d_in[13];
  const float* ffn_w2  = (const float*)d_in[14];
  const float* ffn_b2  = (const float*)d_in[15];
  const float* lnf_s   = (const float*)d_in[16];
  const float* lnf_b   = (const float*)d_in[17];
  float* out = (float*)d_out;

  char*  wsp = (char*)d_ws;
  size_t off = 0;
  auto alloc = [&](size_t bytes) -> char* {
    char* p = wsp + off;
    off += (bytes + 255) & ~(size_t)255;
    return p;
  };
  _Float16* cw  = (_Float16*)alloc((size_t)D_ * KPAT * 2);
  _Float16* qw  = (_Float16*)alloc((size_t)DEPTH * QKVN * D_ * 2);
  _Float16* mw  = (_Float16*)alloc((size_t)DEPTH * D_ * HD * 2);
  _Float16* w1  = (_Float16*)alloc((size_t)DEPTH * MLP_ * D_ * 2);
  _Float16* w2  = (_Float16*)alloc((size_t)DEPTH * D_ * MLP_ * 2);
  float*    t   = (float*)alloc((size_t)MTP * D_ * 4);
  char* zstart = wsp + off;
  _Float16* hb  = (_Float16*)alloc((size_t)MTP * D_ * 2);
  _Float16* peA = (_Float16*)alloc((size_t)NPAT * KPAT * 2);
  _Float16* qb  = (_Float16*)alloc((size_t)32 * SP * DHD * 2);
  _Float16* kb  = (_Float16*)alloc((size_t)32 * SP * DHD * 2);
  _Float16* vT  = (_Float16*)alloc((size_t)32 * DHD * SP * 2);
  _Float16* pr  = (_Float16*)alloc((size_t)32 * SP * SP * 2);
  _Float16* oM  = (_Float16*)alloc((size_t)MTP * HD * 2);
  _Float16* fb  = (_Float16*)alloc((size_t)MTP * MLP_ * 2);
  char* zend = wsp + off;

  {
    const long long n16 = (long long)(zend - zstart) / 16;
    zero_kernel<<<cdiv(n16, 256), 256, 0, stream>>>((uint4*)zstart, n16);
  }

  cvt_kernel<<<cdiv((long long)D_ * KPAT, 256), 256, 0, stream>>>(
      conv_w, cw, (long long)D_ * KPAT);
  for (int l = 0; l < DEPTH; ++l) {
    tr_kernel<<<cdiv((long long)D_ * QKVN, 256), 256, 0, stream>>>(
        qkv_w + (size_t)l * D_ * QKVN, qw + (size_t)l * QKVN * D_, D_, QKVN);
    tr_kernel<<<cdiv((long long)HD * D_, 256), 256, 0, stream>>>(
        merge_w + (size_t)l * HD * D_, mw + (size_t)l * D_ * HD, HD, D_);
    tr_kernel<<<cdiv((long long)D_ * MLP_, 256), 256, 0, stream>>>(
        ffn_w1 + (size_t)l * D_ * MLP_, w1 + (size_t)l * MLP_ * D_, D_, MLP_);
    tr_kernel<<<cdiv((long long)MLP_ * D_, 256), 256, 0, stream>>>(
        ffn_w2 + (size_t)l * MLP_ * D_, w2 + (size_t)l * D_ * MLP_, MLP_, D_);
  }

  patchify_kernel<<<cdiv((long long)NPAT * KPAT, 256), 256, 0, stream>>>(x, peA);
  init_tok_kernel<<<cdiv(BB * D_, 256), 256, 0, stream>>>(cls_tok, pos_emb, t);
  {
    GemmP p{};
    p.A = peA; p.B = cw; p.of = t; p.bias = conv_b; p.aux = pos_emb;
    p.lda = KPAT; p.ldb = KPAT; p.K = KPAT; p.nbRows = D_;
    p.mValid = NPAT; p.ldo = D_;
    gemm_wmma<0><<<dim3(D_ / 256, NPAT / 128, 1), 256, 0, stream>>>(p);
  }

  for (int l = 0; l < DEPTH; ++l) {
    ln_kernel<true><<<MT, 256, 0, stream>>>(t, ln1_s + (size_t)l * D_,
                                            ln1_b + (size_t)l * D_, nullptr, hb);
    {  // QKV
      GemmP p{};
      p.A = hb; p.B = qw + (size_t)l * QKVN * D_;
      p.oh = qb; p.oh2 = kb; p.oh3 = vT;
      p.lda = D_; p.ldb = D_; p.K = D_; p.nbRows = QKVN; p.mValid = MT;
      gemm_wmma<1><<<dim3(QKVN / 256, MTP / 128, 1), 256, 0, stream>>>(p);
    }
    {  // scores over all 32 (b,h)
      GemmP p{};
      p.A = qb; p.B = kb; p.oh = pr;
      p.lda = DHD; p.ldb = DHD; p.K = DHD; p.nbRows = SP;
      p.aBatch = (long long)SP * DHD; p.bBatch = (long long)SP * DHD;
      p.oBatch = (long long)SP * SP;
      gemm_wmma<2><<<dim3(cdiv(SP, 256), SP / 128, 32), 256, 0, stream>>>(p);
    }
    softmax_kernel<<<dim3(NTOK, 32, 1), 256, 0, stream>>>(pr);
    {  // attn @ V
      GemmP p{};
      p.A = pr; p.B = vT; p.oh = oM;
      p.lda = SP; p.ldb = SP; p.K = SP; p.nbRows = DHD;
      p.aBatch = (long long)SP * SP; p.bBatch = (long long)DHD * SP;
      gemm_wmma<3><<<dim3(1, SP / 128, 32), 256, 0, stream>>>(p);
    }
    {  // merge + residual
      GemmP p{};
      p.A = oM; p.B = mw + (size_t)l * D_ * HD; p.of = t;
      p.bias = merge_b + (size_t)l * D_;
      p.lda = HD; p.ldb = HD; p.K = HD; p.nbRows = D_;
      p.mValid = MT; p.ldo = D_;
      gemm_wmma<5><<<dim3(D_ / 256, MTP / 128, 1), 256, 0, stream>>>(p);
    }
    ln_kernel<true><<<MT, 256, 0, stream>>>(t, ln2_s + (size_t)l * D_,
                                            ln2_b + (size_t)l * D_, nullptr, hb);
    {  // FFN1 + GELU
      GemmP p{};
      p.A = hb; p.B = w1 + (size_t)l * MLP_ * D_; p.oh = fb;
      p.bias = ffn_b1 + (size_t)l * MLP_;
      p.lda = D_; p.ldb = D_; p.K = D_; p.nbRows = MLP_;
      p.mValid = MT; p.ldo = MLP_;
      gemm_wmma<4><<<dim3(MLP_ / 256, MTP / 128, 1), 256, 0, stream>>>(p);
    }
    {  // FFN2 + residual
      GemmP p{};
      p.A = fb; p.B = w2 + (size_t)l * D_ * MLP_; p.of = t;
      p.bias = ffn_b2 + (size_t)l * D_;
      p.lda = MLP_; p.ldb = MLP_; p.K = MLP_; p.nbRows = D_;
      p.mValid = MT; p.ldo = D_;
      gemm_wmma<5><<<dim3(D_ / 256, MTP / 128, 1), 256, 0, stream>>>(p);
    }
  }

  ln_kernel<false><<<MT, 256, 0, stream>>>(t, lnf_s, lnf_b, out, nullptr);
}
